// SelfAttention_33380485825042
// MI455X (gfx1250) — matrix operand order repaired and checked by reading, not compiled, not run on hardware
//
#include <hip/hip_runtime.h>
#include <math.h>

#define B_  2
#define S_  2048
#define D_  1024
#define H_  16
#define DH_ 64

typedef __bf16 bf16;
typedef __attribute__((ext_vector_type(16))) __bf16 v16bf;
typedef __attribute__((ext_vector_type(8)))  __bf16 v8bf;
typedef __attribute__((ext_vector_type(8)))  float  v8f;

union V16U { v16bf v; v8bf h[2]; };

// A-operand (16x32 bf16, row-major source): lane<16 -> K 0..15,
// lane>=16 -> K 16..31 (koff = (lane>=16)*8).
__device__ __forceinline__ v16bf ld_a(const bf16* base, int koff) {
  V16U u;
  u.h[0] = *(const v8bf*)(base + 2 * koff);
  u.h[1] = *(const v8bf*)(base + 2 * koff + 8);
  return u.v;
}
// B-operand (32x16 bf16): each lane holds 16 contiguous K values of its column.
__device__ __forceinline__ v16bf ld_b(const bf16* base) {
  V16U u;
  u.h[0] = *(const v8bf*)(base);
  u.h[1] = *(const v8bf*)(base + 8);
  return u.v;
}
__device__ __forceinline__ v8f wmma_bf16(v16bf a, v16bf b, v8f c) {
  return __builtin_amdgcn_wmma_f32_16x16x32_bf16(false, a, false, b, (short)0, c,
                                                 false, false);
}

// ---------------- fp32 -> bf16 conversion ----------------
__global__ __launch_bounds__(256) void cvt_bf16(const float* __restrict__ src,
                                                bf16* __restrict__ dst, int n) {
  int i = blockIdx.x * blockDim.x + threadIdx.x;
  int stride = gridDim.x * blockDim.x;
  for (; i < n; i += stride) dst[i] = (bf16)src[i];
}

// Load one 32-wide K chunk of operands: 2 A row-tiles + 4 B col-tiles.
#define LOAD_CHUNK(dA, aX0, aX1, bX)                                          \
  do {                                                                        \
    aX0 = ld_a(arow0 + (dA), koff);                                           \
    aX1 = ld_a(arow1 + (dA), koff);                                           \
    _Pragma("unroll") for (int t = 0; t < 4; ++t)                             \
      bX[t] = ld_b(wb + (size_t)(e0 + t * 16 + mlane) * D_ + (dA) + hi * 16); \
  } while (0)

// ---------------- QKV projection: qkv = x @ w_in^T + b_in -----------------
// One wave computes a 32x64 strip of C over e in [0,3D). Epilogue scatters
// into q[B,H,S,Dh], k[B,H,S,Dh], vT[B,H,Dh,S] in bf16.
__global__ __launch_bounds__(256) void qkv_gemm(const bf16* __restrict__ xbf,
                                                const bf16* __restrict__ wb,
                                                const float* __restrict__ b_in,
                                                bf16* __restrict__ q,
                                                bf16* __restrict__ k,
                                                bf16* __restrict__ vT) {
  const int lane  = threadIdx.x & 31;
  const int wid   = (blockIdx.x * blockDim.x + threadIdx.x) >> 5;
  const int ST    = (B_ * S_) / 32;          // 128 row strips (32 rows)
  const int nMac  = wid / ST;                // 0..47 (64-wide e strips)
  const int sT    = wid % ST;
  const int s0    = sT * 32;
  const int e0    = nMac * 64;
  const int mlane = lane & 15;
  const int hi    = lane >> 4;
  const int koff  = hi * 8;

  v8f acc[2][4] = {};
  const bf16* arow0 = xbf + (size_t)(s0 + mlane) * D_;
  const bf16* arow1 = arow0 + (size_t)16 * D_;

  v16bf aA0, aA1, bA[4];       // ping buffer (chunk d0)
  v16bf aB0, aB1, bB[4];       // pong buffer (chunk d0+32)
  LOAD_CHUNK(0, aA0, aA1, bA);
  for (int d0 = 0; d0 < D_; d0 += 64) {
    __builtin_prefetch(arow0 + d0 + 256, 0, 3);
    LOAD_CHUNK(d0 + 32, aB0, aB1, bB);
#pragma unroll
    for (int t = 0; t < 4; ++t) acc[0][t] = wmma_bf16(aA0, bA[t], acc[0][t]);
#pragma unroll
    for (int t = 0; t < 4; ++t) acc[1][t] = wmma_bf16(aA1, bA[t], acc[1][t]);
    if (d0 + 64 < D_) LOAD_CHUNK(d0 + 64, aA0, aA1, bA);
#pragma unroll
    for (int t = 0; t < 4; ++t) acc[0][t] = wmma_bf16(aB0, bB[t], acc[0][t]);
#pragma unroll
    for (int t = 0; t < 4; ++t) acc[1][t] = wmma_bf16(aB1, bB[t], acc[1][t]);
  }

  const int bidx = s0 >> 11;            // S_ == 2048
  const int sl0  = s0 & (S_ - 1);
#pragma unroll
  for (int rt = 0; rt < 2; ++rt) {
#pragma unroll
    for (int t = 0; t < 4; ++t) {
      int   e    = e0 + t * 16 + mlane;
      float bias = b_in[e];
#pragma unroll
      for (int r = 0; r < 8; ++r) {
        int   sl  = sl0 + rt * 16 + r + hi * 8;
        float val = acc[rt][t][r] + bias;
        if (e < D_) {
          int h = e >> 6, dh = e & 63;
          q[(((size_t)bidx * H_ + h) * S_ + sl) * DH_ + dh] = (bf16)val;
        } else if (e < 2 * D_) {
          int e2 = e - D_;
          int h = e2 >> 6, dh = e2 & 63;
          k[(((size_t)bidx * H_ + h) * S_ + sl) * DH_ + dh] = (bf16)val;
        } else {
          int e2 = e - 2 * D_;
          int h = e2 >> 6, dh = e2 & 63;
          vT[(((size_t)bidx * H_ + h) * DH_ + dh) * S_ + sl] = (bf16)val;
        }
      }
    }
  }
}

// ---------------- Flash attention (causal), one wave per 16 query rows ----
__global__ __launch_bounds__(256) void attn(const bf16* __restrict__ q,
                                            const bf16* __restrict__ kmat,
                                            const bf16* __restrict__ vT,
                                            bf16* __restrict__ obf) {
  __shared__ bf16 p_lds[8][16][32];     // per-wave P tile, C-layout -> A-layout
  const int lane  = threadIdx.x & 31;
  const int w     = threadIdx.x >> 5;
  const int wid   = blockIdx.x * (blockDim.x >> 5) + w;
  const int QT    = S_ / 16;            // 128
  const int bh    = wid / QT;
  const int qt    = wid % QT;
  const int b     = bh / H_, h = bh % H_;
  const int q0    = qt * 16;
  const int mlane = lane & 15;
  const int hi    = lane >> 4;

  const bf16* qbase = q    + (((size_t)b * H_ + h) * S_) * DH_;
  const bf16* kbase = kmat + (((size_t)b * H_ + h) * S_) * DH_;
  const bf16* vbase = vT   + (((size_t)b * H_ + h) * DH_) * S_;

  const bf16* qrow = qbase + (size_t)(q0 + mlane) * DH_;
  v16bf aq0 = ld_a(qrow + 0,  hi * 8);   // d 0..31
  v16bf aq1 = ld_a(qrow + 32, hi * 8);   // d 32..63

  float mrow[8], lrow[8];
#pragma unroll
  for (int r = 0; r < 8; ++r) { mrow[r] = -__builtin_inff(); lrow[r] = 0.f; }
  v8f accO[4] = {};

  const float sscale = 0.125f;           // 1/sqrt(Dh)
  const int kend = q0 + 16;
  for (int k0 = 0; k0 < kend; k0 += 32) {
    const bool has2 = (k0 + 16) < kend;
    // ---- issue ALL global loads of this k-step up front ----
    v16bf bv[4];
#pragma unroll
    for (int t = 0; t < 4; ++t)
      bv[t] = ld_b(vbase + (size_t)(t * 16 + mlane) * S_ + k0 + hi * 16);
    const bf16* krow0 = kbase + (size_t)(k0 + mlane) * DH_ + hi * 16;
    v16bf bk00 = ld_b(krow0), bk01 = ld_b(krow0 + 32);
    v16bf bk10, bk11;
    if (has2) {
      const bf16* krow1 = kbase + (size_t)(k0 + 16 + mlane) * DH_ + hi * 16;
      bk10 = ld_b(krow1);
      bk11 = ld_b(krow1 + 32);
    }
    // ---- scores ----
    float sv0[8], sv1[8], p0[8], p1[8];
    {
      v8f s = {};
      s = wmma_bf16(aq0, bk00, s);
      s = wmma_bf16(aq1, bk01, s);
#pragma unroll
      for (int r = 0; r < 8; ++r) {
        int sq = q0 + r + hi * 8, sk = k0 + mlane;
        float f = s[r] * sscale;
        sv0[r] = (sk > sq) ? -__builtin_inff() : f;
      }
    }
    if (has2) {
      v8f s = {};
      s = wmma_bf16(aq0, bk10, s);
      s = wmma_bf16(aq1, bk11, s);
#pragma unroll
      for (int r = 0; r < 8; ++r) {
        int sq = q0 + r + hi * 8, sk = k0 + 16 + mlane;
        float f = s[r] * sscale;
        sv1[r] = (sk > sq) ? -__builtin_inff() : f;
      }
    } else {
#pragma unroll
      for (int r = 0; r < 8; ++r) sv1[r] = -__builtin_inff();
    }
    // ---- online softmax update (row reductions across 16-lane halves) ----
#pragma unroll
    for (int r = 0; r < 8; ++r) {
      float tm = fmaxf(sv0[r], sv1[r]);
      tm = fmaxf(tm, __shfl_xor(tm, 1, 32));
      tm = fmaxf(tm, __shfl_xor(tm, 2, 32));
      tm = fmaxf(tm, __shfl_xor(tm, 4, 32));
      tm = fmaxf(tm, __shfl_xor(tm, 8, 32));
      float mnew = fmaxf(mrow[r], tm);
      float corr = __expf(mrow[r] - mnew);
      mrow[r] = mnew;
      float a0 = __expf(sv0[r] - mnew);
      float a1 = __expf(sv1[r] - mnew);
      p0[r] = a0; p1[r] = a1;
      float ps = a0 + a1;
      ps += __shfl_xor(ps, 1, 32);
      ps += __shfl_xor(ps, 2, 32);
      ps += __shfl_xor(ps, 4, 32);
      ps += __shfl_xor(ps, 8, 32);
      lrow[r] = lrow[r] * corr + ps;
#pragma unroll
      for (int t = 0; t < 4; ++t) accO[t][r] = accO[t][r] * corr;
    }
    // ---- C-layout P -> LDS -> A-layout (same-wave DS ops are in-order) ----
#pragma unroll
    for (int r = 0; r < 8; ++r) {
      int row = r + hi * 8;
      p_lds[w][row][mlane]      = (bf16)p0[r];
      p_lds[w][row][16 + mlane] = (bf16)p1[r];
    }
    v16bf ap;
    {
      V16U u;
      u.h[0] = *(const v8bf*)&p_lds[w][mlane][hi * 16];
      u.h[1] = *(const v8bf*)&p_lds[w][mlane][hi * 16 + 8];
      ap = u.v;
    }
    // ---- O += P(16x32) @ V(32x64); V operands already in flight ----
#pragma unroll
    for (int t = 0; t < 4; ++t) accO[t] = wmma_bf16(ap, bv[t], accO[t]);
  }
  // epilogue: normalize and write bf16 O in [B,S,D] layout for out-proj
#pragma unroll
  for (int r = 0; r < 8; ++r) {
    float inv = 1.0f / lrow[r];
    int srow  = q0 + r + hi * 8;
    size_t base = ((size_t)b * S_ + srow) * D_ + h * DH_;
#pragma unroll
    for (int t = 0; t < 4; ++t)
      obf[base + t * 16 + mlane] = (bf16)(accO[t][r] * inv);
  }
}

// ---------------- Output projection: out = o @ w_out^T + b_out (fp32) -----
__global__ __launch_bounds__(256) void out_gemm(const bf16* __restrict__ obf,
                                                const bf16* __restrict__ wb,
                                                const float* __restrict__ b_out,
                                                float* __restrict__ out) {
  const int lane  = threadIdx.x & 31;
  const int wid   = (blockIdx.x * blockDim.x + threadIdx.x) >> 5;
  const int ST    = (B_ * S_) / 32;     // 128 row strips
  const int nMac  = wid / ST;           // 0..15
  const int sT    = wid % ST;
  const int s0    = sT * 32;
  const int e0    = nMac * 64;
  const int mlane = lane & 15;
  const int hi    = lane >> 4;
  const int koff  = hi * 8;

  v8f acc[2][4] = {};
  const bf16* arow0 = obf + (size_t)(s0 + mlane) * D_;
  const bf16* arow1 = arow0 + (size_t)16 * D_;

  v16bf aA0, aA1, bA[4];
  v16bf aB0, aB1, bB[4];
  LOAD_CHUNK(0, aA0, aA1, bA);
  for (int d0 = 0; d0 < D_; d0 += 64) {
    __builtin_prefetch(arow0 + d0 + 256, 0, 3);
    LOAD_CHUNK(d0 + 32, aB0, aB1, bB);
#pragma unroll
    for (int t = 0; t < 4; ++t) acc[0][t] = wmma_bf16(aA0, bA[t], acc[0][t]);
#pragma unroll
    for (int t = 0; t < 4; ++t) acc[1][t] = wmma_bf16(aA1, bA[t], acc[1][t]);
    if (d0 + 64 < D_) LOAD_CHUNK(d0 + 64, aA0, aA1, bA);
#pragma unroll
    for (int t = 0; t < 4; ++t) acc[0][t] = wmma_bf16(aB0, bB[t], acc[0][t]);
#pragma unroll
    for (int t = 0; t < 4; ++t) acc[1][t] = wmma_bf16(aB1, bB[t], acc[1][t]);
  }
#pragma unroll
  for (int rt = 0; rt < 2; ++rt) {
#pragma unroll
    for (int t = 0; t < 4; ++t) {
      int   e    = e0 + t * 16 + mlane;
      float bias = b_out[e];
#pragma unroll
      for (int r = 0; r < 8; ++r) {
        size_t sg = (size_t)(s0 + rt * 16 + r + hi * 8);
        out[sg * D_ + e] = acc[rt][t][r] + bias;
      }
    }
  }
}

extern "C" void kernel_launch(void* const* d_in, const int* in_sizes, int n_in,
                              void* d_out, int out_size, void* d_ws, size_t ws_size,
                              hipStream_t stream) {
  const float* x     = (const float*)d_in[0];
  const float* w_in  = (const float*)d_in[1];
  const float* b_in  = (const float*)d_in[2];
  const float* w_out = (const float*)d_in[3];
  const float* b_out = (const float*)d_in[4];
  float* out = (float*)d_out;

  char* ws = (char*)d_ws;
  const size_t MB = 1ull << 20;
  bf16* xbf  = (bf16*)(ws + 0 * MB);    //  8 MB: x in bf16        [B,S,D]
  bf16* wibf = (bf16*)(ws + 8 * MB);    //  6 MB: w_in bf16        [3D,D]
  bf16* wobf = (bf16*)(ws + 14 * MB);   //  2 MB: w_out bf16       [D,D]
  bf16* qb   = (bf16*)(ws + 16 * MB);   //  8 MB: Q bf16           [B,H,S,Dh]
  bf16* kb   = (bf16*)(ws + 24 * MB);   //  8 MB: K bf16           [B,H,S,Dh]
  bf16* vtb  = (bf16*)(ws + 32 * MB);   //  8 MB: V^T bf16         [B,H,Dh,S]
  bf16* ob   = (bf16*)(ws + 40 * MB);   //  8 MB: attn out bf16    [B,S,D]

  cvt_bf16<<<4096, 256, 0, stream>>>(x,     xbf,  B_ * S_ * D_);
  cvt_bf16<<<3072, 256, 0, stream>>>(w_in,  wibf, 3 * D_ * D_);
  cvt_bf16<<<1024, 256, 0, stream>>>(w_out, wobf, D_ * D_);

  // 48 e-strips x 128 s-strips = 6144 waves; 8 waves/block
  qkv_gemm<<<6144 / 8, 256, 0, stream>>>(xbf, wibf, b_in, qb, kb, vtb);
  // B*H*(S/16) = 4096 waves
  attn<<<4096 / 8, 256, 0, stream>>>(qb, kb, vtb, ob);
  // 16 e-strips x 128 s-strips = 2048 waves
  out_gemm<<<2048 / 8, 256, 0, stream>>>(ob, wobf, b_out, out);
}
